// SelfAttention_66056597013109
// MI455X (gfx1250) — compile-verified, matrix operand order checked by reading, not run on hardware
//
#include <hip/hip_runtime.h>
#include <hip/hip_bf16.h>

// ---------------------------------------------------------------------------
// Phi-2 style self-attention for MI455X (gfx1250, wave32, WMMA).
//   SEQ=2048, N_EMBD=2560, N_HEAD=32, HEAD_DIM=80, ROTARY_DIM=32
// Pipeline:
//   cast(hidden,Wqkv)->f16 -> QKV GEMM(+bias, f32 out) -> RoPE(f32, in place)
//   -> cast qkv->f16 -> flash attention (f16 operands, f32 accum, f16 out)
//   -> out GEMM(+bias) -> f32 result.
// GEMM stages use v_wmma_f32_16x16x32_f16 with double-buffered LDS staging;
// staging uses GLOBAL_LOAD_ASYNC_TO_LDS_B128 when the builtin is available.
// ---------------------------------------------------------------------------

typedef __attribute__((ext_vector_type(16))) _Float16 v16h;
typedef __attribute__((ext_vector_type(8)))  _Float16 v8h;
typedef __attribute__((ext_vector_type(8)))  float    v8f;

#define SEQ     2048
#define NEMBD   2560
#define NHEAD   32
#define HDIM    80
#define OPSZ    7680   // 3 * NEMBD

#if __has_builtin(__builtin_amdgcn_global_load_async_to_lds_b128)
#define HAVE_ASYNC_LDS 1
#else
#define HAVE_ASYNC_LDS 0
#endif

typedef int v4i __attribute__((vector_size(16)));
typedef __attribute__((address_space(1))) v4i gv4i_t;   // global int4
typedef __attribute__((address_space(3))) v4i lv4i_t;   // LDS int4

// Copy 16 bytes global->LDS; async (ASYNCcnt) when available.
static __device__ __forceinline__ void cp16_g2l(const _Float16* g, _Float16* l) {
#if HAVE_ASYNC_LDS
  __builtin_amdgcn_global_load_async_to_lds_b128((gv4i_t*)g, (lv4i_t*)l, 0, 0);
#else
  *(v8h*)l = *(const v8h*)g;
#endif
}

static __device__ __forceinline__ void async_wait_all() {
#if HAVE_ASYNC_LDS
#if __has_builtin(__builtin_amdgcn_s_wait_asynccnt)
  __builtin_amdgcn_s_wait_asynccnt(0);
#else
  asm volatile("s_wait_asynccnt 0x0" ::: "memory");
#endif
#endif
}

static __device__ __forceinline__ v8f wmma_f16(v16h a, v16h b, v8f c) {
  return __builtin_amdgcn_wmma_f32_16x16x32_f16(false, a, false, b,
                                                (short)0, c, false, false);
}

// Load a 16x32 f16 fragment (A-layout == B-layout under CDNA5 wave32 WMMA):
//   lane L holds row (row0 + L%16); lanes 0-15 carry K {kbase+0..7, kbase+16..23},
//   lanes 16-31 carry K {kbase+8..15, kbase+24..31}.
// Row stride (halves) must be a multiple of 8 so v8h loads stay 16B aligned.
static __device__ __forceinline__ v16h load_frag16(const _Float16* base,
                                                   int stride, int row0, int kbase) {
  const int l = threadIdx.x & 31;
  const _Float16* p = base + (size_t)(row0 + (l & 15)) * stride + kbase + ((l >> 4) << 3);
  v8h lo = *(const v8h*)p;
  v8h hi = *(const v8h*)(p + 16);
  return __builtin_shufflevector(lo, hi, 0,1,2,3,4,5,6,7,8,9,10,11,12,13,14,15);
}

// Convert 8 consecutive f32 -> v8h.
static __device__ __forceinline__ v8h cvt8(const float* p) {
  const float4 f0 = *(const float4*)p;
  const float4 f1 = *(const float4*)(p + 4);
  v8h r = { (_Float16)f0.x, (_Float16)f0.y, (_Float16)f0.z, (_Float16)f0.w,
            (_Float16)f1.x, (_Float16)f1.y, (_Float16)f1.z, (_Float16)f1.w };
  return r;
}

// ---------------------------------------------------------------------------
// Elementwise f32 -> f16 cast, 8 elems per thread per grid-stride step.
// ---------------------------------------------------------------------------
__global__ __launch_bounds__(256)
void cast_f32_to_f16(const float* __restrict__ src, _Float16* __restrict__ dst,
                     int n8) {
  for (size_t i = (size_t)blockIdx.x * 256 + threadIdx.x; i < (size_t)n8;
       i += (size_t)gridDim.x * 256)
    *(v8h*)(dst + i * 8) = cvt8(src + i * 8);
}

// ---------------------------------------------------------------------------
// C[M,N] = A[M,K] * W[N,K]^T + bias[N]; A, W row-major f16 (K contiguous).
// Block: 256 threads = 8 waves; tile 128(M) x 64(N); K in steps of 32,
// double-buffered in LDS with async global->LDS staging.
// ---------------------------------------------------------------------------
__global__ __launch_bounds__(256)
void gemm_nt_bias(const _Float16* __restrict__ A, const _Float16* __restrict__ W,
                  const float* __restrict__ bias, float* __restrict__ C,
                  int M, int N, int K) {
  __shared__ alignas(16) _Float16 As[2][128][40];  // 32 K + 8 pad
  __shared__ alignas(16) _Float16 Bs[2][64][40];

  const int tid  = threadIdx.x;
  const int row0 = blockIdx.y * 128;
  const int col0 = blockIdx.x * 64;
  const int wave = tid >> 5, lane = tid & 31;
  const int wm = wave >> 1, wn = wave & 1;

  v8f acc[2][2] = {};

  // Stage one 32-wide K slab into buffer `b` (16B chunks, async when possible).
  auto stage = [&](int b, int kk) {
    #pragma unroll
    for (int i = tid; i < 512; i += 256) {           // A: 128 rows x 4 chunks
      int r = i >> 2, c = (i & 3) * 8;
      cp16_g2l(A + (size_t)(row0 + r) * K + kk + c, &As[b][r][c]);
    }
    {
      int i = tid;                                    // B: 64 rows x 4 chunks
      int r = i >> 2, c = (i & 3) * 8;
      cp16_g2l(W + (size_t)(col0 + r) * K + kk + c, &Bs[b][r][c]);
    }
  };

  stage(0, 0);
  int buf = 0;
  for (int kk = 0; kk < K; kk += 32) {
    async_wait_all();
    __syncthreads();                 // staged slab visible; prev reads retired
    if (kk + 32 < K) stage(buf ^ 1, kk + 32);

    v16h a0 = load_frag16(&As[buf][0][0], 40, wm * 32,      0);
    v16h a1 = load_frag16(&As[buf][0][0], 40, wm * 32 + 16, 0);
    v16h b0 = load_frag16(&Bs[buf][0][0], 40, wn * 32,      0);
    v16h b1 = load_frag16(&Bs[buf][0][0], 40, wn * 32 + 16, 0);
    acc[0][0] = wmma_f16(a0, b0, acc[0][0]);
    acc[0][1] = wmma_f16(a0, b1, acc[0][1]);
    acc[1][0] = wmma_f16(a1, b0, acc[1][0]);
    acc[1][1] = wmma_f16(a1, b1, acc[1][1]);
    buf ^= 1;
  }

  // C/D layout: lane holds (M = vr + 8*(lane/16), N = lane%16)
  const int hw = lane >> 4, cl = lane & 15;
  for (int i = 0; i < 2; ++i)
    for (int j = 0; j < 2; ++j) {
      int col = col0 + wn * 32 + j * 16 + cl;
      float bv = bias[col];
      #pragma unroll
      for (int vr = 0; vr < 8; ++vr) {
        int row = row0 + wm * 32 + i * 16 + 8 * hw + vr;
        C[(size_t)row * N + col] = acc[i][j][vr] + bv;
      }
    }
}

// ---------------------------------------------------------------------------
// Partial rotary (dim 32) applied in-place (f32) to q and k halves of QKV.
// Reference: out = x*cos + rotate_half(x)*sin. Thread owns a (d,d+16) pair.
// ---------------------------------------------------------------------------
__global__ __launch_bounds__(256)
void rope_kernel(float* __restrict__ qkv, const float* __restrict__ cosT,
                 const float* __restrict__ sinT) {
  const int t = blockIdx.x;
  #pragma unroll
  for (int it = 0; it < 2; ++it) {
    int idx = threadIdx.x + it * 256;   // 0..511 -> 32 heads x 16 pairs
    int hh = idx >> 4, d = idx & 15;
    float c0 = cosT[t * 32 + d],      s0 = sinT[t * 32 + d];
    float c1 = cosT[t * 32 + d + 16], s1 = sinT[t * 32 + d + 16];
    float* qr = qkv + (size_t)t * OPSZ + hh * HDIM;
    float x0 = qr[d], x1 = qr[d + 16];
    qr[d]      = x0 * c0 + x1 * s0;
    qr[d + 16] = x1 * c1 + x0 * s1;
    float* kr = qr + NEMBD;
    float y0 = kr[d], y1 = kr[d + 16];
    kr[d]      = y0 * c0 + y1 * s0;
    kr[d + 16] = y1 * c1 + y0 * s1;
  }
}

// ---------------------------------------------------------------------------
// Flash attention over f16 qkv. Block = (head, 64-query tile), 4 waves,
// 16 query rows each. Head dim 80 padded to 96 (3 K-steps of 32); pad
// columns zeroed once, live columns staged with async b128 copies.
// ---------------------------------------------------------------------------
__global__ __launch_bounds__(128)
void attn_kernel(const _Float16* __restrict__ qkv, _Float16* __restrict__ ctx) {
  __shared__ alignas(16) _Float16 Qs[64][104];   // 96 cols (80 + zero pad) + 8
  __shared__ alignas(16) _Float16 Ks[32][104];
  __shared__ alignas(16) _Float16 Vt[80][40];    // V transposed: [dim][key]
  __shared__ alignas(16) _Float16 Ps[4][16][40]; // per-wave P tile, A layout

  const int h  = blockIdx.x;
  const int qt = blockIdx.y;
  const int qbase = qt * 64;
  const int tid  = threadIdx.x;
  const int w    = tid >> 5, lane = tid & 31;
  const int hw   = lane >> 4, cl = lane & 15;

  // Zero pad columns 80..95 (never overwritten by async copies).
  for (int i = tid; i < 64 * 16; i += 128) Qs[i >> 4][80 + (i & 15)] = (_Float16)0.f;
  for (int i = tid; i < 32 * 16; i += 128) Ks[i >> 4][80 + (i & 15)] = (_Float16)0.f;

  // Stage Q tile (64 rows x 80 halves = 640 x 16B chunks).
  for (int i = tid; i < 640; i += 128) {
    int r = i / 10, c = (i % 10) * 8;
    cp16_g2l(qkv + (size_t)(qbase + r) * OPSZ + h * HDIM + c, &Qs[r][c]);
  }
  async_wait_all();
  __syncthreads();

  v16h qa[3];
  #pragma unroll
  for (int ks = 0; ks < 3; ++ks)
    qa[ks] = load_frag16(&Qs[0][0], 104, w * 16, ks * 32);

  v8f acc[5] = {};
  float mrow[8], lrow[8];
  #pragma unroll
  for (int vr = 0; vr < 8; ++vr) { mrow[vr] = -1e30f; lrow[vr] = 0.0f; }

  const float scale = 0.11180339887498949f;  // 1/sqrt(80)
  const int nkt = qt * 2 + 2;                // causal: keys up to qbase+63

  for (int kt = 0; kt < nkt; ++kt) {
    __syncthreads();  // prior K/V reads retired before restaging
    for (int i = tid; i < 320; i += 128) {          // K: 32 rows x 10 chunks
      int r = i / 10, c = (i % 10) * 8;
      cp16_g2l(qkv + (size_t)(kt * 32 + r) * OPSZ + NEMBD + h * HDIM + c, &Ks[r][c]);
    }
    for (int i = tid; i < 32 * 80; i += 128) {      // V transposed (manual)
      int r = i / 80, c = i % 80;
      Vt[c][r] = qkv[(size_t)(kt * 32 + r) * OPSZ + 2 * NEMBD + h * HDIM + c];
    }
    async_wait_all();
    __syncthreads();

    // S tile: 16 q rows x 32 keys per wave.
    v8f s0 = {}, s1 = {};
    #pragma unroll
    for (int ks = 0; ks < 3; ++ks) {
      v16h b0 = load_frag16(&Ks[0][0], 104, 0,  ks * 32);
      v16h b1 = load_frag16(&Ks[0][0], 104, 16, ks * 32);
      s0 = wmma_f16(qa[ks], b0, s0);
      s1 = wmma_f16(qa[ks], b1, s1);
    }

    // Online softmax; C layout => lane holds rows {8*hw + 0..7} at cols cl/cl+16.
    #pragma unroll
    for (int vr = 0; vr < 8; ++vr) {
      int qrow = qbase + w * 16 + 8 * hw + vr;
      float e0 = s0[vr] * scale; if (kt * 32 + cl      > qrow) e0 = -1e30f;
      float e1 = s1[vr] * scale; if (kt * 32 + 16 + cl > qrow) e1 = -1e30f;
      float rmax = fmaxf(e0, e1);
      #pragma unroll
      for (int mk = 1; mk < 16; mk <<= 1)
        rmax = fmaxf(rmax, __shfl_xor(rmax, mk, 32));
      float mnew = fmaxf(mrow[vr], rmax);
      float al   = __expf(mrow[vr] - mnew);
      float p0   = __expf(e0 - mnew);
      float p1   = __expf(e1 - mnew);
      float rs   = p0 + p1;
      #pragma unroll
      for (int mk = 1; mk < 16; mk <<= 1)
        rs += __shfl_xor(rs, mk, 32);
      lrow[vr] = lrow[vr] * al + rs;
      mrow[vr] = mnew;
      #pragma unroll
      for (int j = 0; j < 5; ++j) acc[j][vr] *= al;
      Ps[w][8 * hw + vr][cl]      = (_Float16)p0;
      Ps[w][8 * hw + vr][16 + cl] = (_Float16)p1;
    }
    // Intra-wave LDS RAW (other lanes' P stores) before fragment reload.
    asm volatile("s_wait_dscnt 0" ::: "memory");

    v16h pa = load_frag16(&Ps[w][0][0], 40, 0, 0);
    #pragma unroll
    for (int j = 0; j < 5; ++j) {
      v16h vb = load_frag16(&Vt[0][0], 40, j * 16, 0);  // B: rows=dim, K=32 keys
      acc[j] = wmma_f16(pa, vb, acc[j]);
    }
  }

  #pragma unroll
  for (int j = 0; j < 5; ++j)
    #pragma unroll
    for (int vr = 0; vr < 8; ++vr) {
      int row = qbase + w * 16 + 8 * hw + vr;
      int c   = h * HDIM + j * 16 + cl;
      ctx[(size_t)row * NEMBD + c] = (_Float16)(acc[j][vr] / lrow[vr]);
    }
}

// ---------------------------------------------------------------------------
extern "C" void kernel_launch(void* const* d_in, const int* in_sizes, int n_in,
                              void* d_out, int out_size, void* d_ws, size_t ws_size,
                              hipStream_t stream) {
  const float* hidden = (const float*)d_in[0];
  const float* Wqkv_w = (const float*)d_in[1];
  const float* Wqkv_b = (const float*)d_in[2];
  const float* out_w  = (const float*)d_in[3];
  const float* out_b  = (const float*)d_in[4];
  const float* cosT   = (const float*)d_in[5];
  const float* sinT   = (const float*)d_in[6];
  float* out = (float*)d_out;

  // Workspace layout (phase-disjoint regions reuse the tail):
  //   [0)                qkv_f32  : SEQ*OPSZ f32  (62.9 MB)
  //   [A)                qkv16    : SEQ*OPSZ f16  (31.5 MB)
  //   [B) phase 1:       h16 | Wq16        (10.5 + 39.3 MB)
  //       phase 2 (after GEMM1): Wo16 | ctx16 (13.1 + 10.5 MB)
  char* ws = (char*)d_ws;
  float*     qkv32 = (float*)ws;
  _Float16*  qkv16 = (_Float16*)(ws + (size_t)SEQ * OPSZ * 4);
  char*      phase = ws + (size_t)SEQ * OPSZ * 4 + (size_t)SEQ * OPSZ * 2;
  _Float16*  h16   = (_Float16*)phase;
  _Float16*  Wq16  = (_Float16*)(phase + (size_t)SEQ * NEMBD * 2);
  _Float16*  Wo16  = (_Float16*)phase;                                   // reuse
  _Float16*  ctx16 = (_Float16*)(phase + (size_t)NEMBD * NEMBD * 2);     // reuse

  const int n8_h   = SEQ * NEMBD / 8;
  const int n8_wq  = OPSZ * NEMBD / 8;
  const int n8_wo  = NEMBD * NEMBD / 8;
  const int n8_qkv = SEQ * OPSZ / 8;

  // 1) cast operands to f16
  cast_f32_to_f16<<<(n8_h  + 255) / 256, 256, 0, stream>>>(hidden, h16,  n8_h);
  cast_f32_to_f16<<<(n8_wq + 255) / 256, 256, 0, stream>>>(Wqkv_w, Wq16, n8_wq);
  // 2) QKV = hidden @ Wqkv^T + b   (f32 output for exact-precision RoPE)
  gemm_nt_bias<<<dim3(OPSZ / 64, SEQ / 128), 256, 0, stream>>>(
      h16, Wq16, Wqkv_b, qkv32, SEQ, OPSZ, NEMBD);
  // 3) partial RoPE in f32, then cast full qkv to f16
  rope_kernel<<<SEQ, 256, 0, stream>>>(qkv32, cosT, sinT);
  cast_f32_to_f16<<<(n8_qkv + 255) / 256, 256, 0, stream>>>(qkv32, qkv16, n8_qkv);
  // 4) out_w cast (reuses phase region freed by GEMM1)
  cast_f32_to_f16<<<(n8_wo + 255) / 256, 256, 0, stream>>>(out_w, Wo16, n8_wo);
  // 5) causal flash attention -> ctx16
  attn_kernel<<<dim3(NHEAD, SEQ / 64), 128, 0, stream>>>(qkv16, ctx16);
  // 6) out = ctx @ out_w^T + out_b
  gemm_nt_bias<<<dim3(NEMBD / 64, SEQ / 128), 256, 0, stream>>>(
      ctx16, Wo16, out_b, out, SEQ, NEMBD, NEMBD);
}
